// AnyPrecisionLinear_87814901334605
// MI455X (gfx1250) — matrix operand order repaired and checked by reading, not compiled, not run on hardware
//
#include <hip/hip_runtime.h>
#include <hip/hip_bf16.h>
#include <stdint.h>

// ---------------------------------------------------------------------------
// AnyPrecision 4-bit linear for MI455X (gfx1250, wave32, WMMA).
//
// GEMM: M=8192, N=4096, K=4096 -> 275 GFLOP, compute bound on bf16 WMMA.
// Pass 1 : dequant qweight bit-planes + LUT -> bf16 W image (32 MiB in d_ws)
// Pass 1b: convert x fp32 -> bf16 image (64 MiB in d_ws)
// Pass 2 : double-buffered tiled GEMM, tiles staged with
//          GLOBAL_LOAD_ASYNC_TO_LDS_B128 (ASYNCcnt) when available.
// ---------------------------------------------------------------------------

typedef __attribute__((ext_vector_type(16))) __bf16 v16bf;
typedef __attribute__((ext_vector_type(8)))  __bf16 v8bf;
typedef __attribute__((ext_vector_type(8)))  float  v8f;
typedef int v4i32 __attribute__((vector_size(16))); // matches builtin's V4i

#define K_TOT 4096
#define N_TOT 4096
#define M_TOT 8192
#define NWORDS (K_TOT / 32)
#define PLANE_STRIDE (N_TOT * NWORDS)

#define BM 256
#define BN 128
#define BK 64
#define LDK 72                      // padded LDS row stride (bf16): conflict-free
#define BUF_ELEMS ((BM + BN) * LDK) // one ping-pong buffer, bf16 elems

#if defined(__HIP_DEVICE_COMPILE__) && \
    __has_builtin(__builtin_amdgcn_global_load_async_to_lds_b128)
#define HAVE_ASYNC 1
#else
#define HAVE_ASYNC 0
#endif

__device__ __forceinline__ void wait_async_all() {
#if __has_builtin(__builtin_amdgcn_s_wait_asynccnt)
  __builtin_amdgcn_s_wait_asynccnt(0);
#else
  asm volatile("s_wait_asynccnt 0x0" ::: "memory");
#endif
}

// Copy one 16B chunk global->LDS (async if available).
__device__ __forceinline__ void copy16_to_lds(const uint32_t* __restrict__ src,
                                              __bf16* dst) {
#if HAVE_ASYNC
  __builtin_amdgcn_global_load_async_to_lds_b128(
      (__attribute__((address_space(1))) v4i32*)(uintptr_t)src,
      (__attribute__((address_space(3))) v4i32*)(uint32_t)(uintptr_t)dst,
      /*offset=*/0, /*cpol=*/0);
#else
  *(uint4*)dst = *(const uint4*)src;
#endif
}

// ---------------------------------------------------------------------------
// Pass 1: dequantize 4-bit LUT weights into bf16 (O x In) row-major image.
// ---------------------------------------------------------------------------
__global__ __launch_bounds__(256)
void anyprec_dequant_kernel(const int* __restrict__ q,
                            const float* __restrict__ lut,
                            uint32_t* __restrict__ W) {
  const int g    = blockIdx.x * 256 + threadIdx.x;
  const int o    = g >> 7;
  const int w    = g & (NWORDS - 1);
  const int lane = threadIdx.x & 31;

  const size_t base = (size_t)o * NWORDS + w;
  const uint32_t q0 = (uint32_t)q[base];
  const uint32_t q1 = (uint32_t)q[base + (size_t)PLANE_STRIDE];
  const uint32_t q2 = (uint32_t)q[base + (size_t)2 * PLANE_STRIDE];
  const uint32_t q3 = (uint32_t)q[base + (size_t)3 * PLANE_STRIDE];

  const float lv = lut[o * 16 + (lane & 15)];

  uint32_t packed[16];
#pragma unroll
  for (int i = 0; i < 16; ++i) {
    const int i0 = 2 * i, i1 = 2 * i + 1;
    const int idx0 = (int)(((q0 >> i0) & 1u) | (((q1 >> i0) & 1u) << 1) |
                           (((q2 >> i0) & 1u) << 2) | (((q3 >> i0) & 1u) << 3));
    const int idx1 = (int)(((q0 >> i1) & 1u) | (((q1 >> i1) & 1u) << 1) |
                           (((q2 >> i1) & 1u) << 2) | (((q3 >> i1) & 1u) << 3));
    const float f0 = __shfl(lv, idx0, 16);
    const float f1 = __shfl(lv, idx1, 16);
    const __bf16 b0 = (__bf16)f0;
    const __bf16 b1 = (__bf16)f1;
    packed[i] = (uint32_t)__builtin_bit_cast(unsigned short, b0) |
                ((uint32_t)__builtin_bit_cast(unsigned short, b1) << 16);
  }

  uint4* dst = (uint4*)(W + (size_t)o * (K_TOT / 2) + (size_t)w * 16);
#pragma unroll
  for (int j = 0; j < 4; ++j)
    dst[j] = make_uint4(packed[4 * j], packed[4 * j + 1],
                        packed[4 * j + 2], packed[4 * j + 3]);
}

// ---------------------------------------------------------------------------
// Pass 1b: x fp32 -> bf16 image (8 elems / thread, 16B stores).
// ---------------------------------------------------------------------------
__global__ __launch_bounds__(256)
void x_to_bf16_kernel(const float* __restrict__ x, uint32_t* __restrict__ Xb) {
  const size_t i = ((size_t)blockIdx.x * 256 + threadIdx.x) * 8;
  const float4 f0 = *(const float4*)(x + i);
  const float4 f1 = *(const float4*)(x + i + 4);
  v8bf b;
  b[0] = (__bf16)f0.x; b[1] = (__bf16)f0.y;
  b[2] = (__bf16)f0.z; b[3] = (__bf16)f0.w;
  b[4] = (__bf16)f1.x; b[5] = (__bf16)f1.y;
  b[6] = (__bf16)f1.z; b[7] = (__bf16)f1.w;
  *(v8bf*)(Xb + i / 2) = b;
}

// ---------------------------------------------------------------------------
// Tile staging: issue A (256x64) + B (128x64) bf16 copies into one LDS buffer.
// 8 + 4 x 16B per thread, fully coalesced.
// ---------------------------------------------------------------------------
__device__ __forceinline__ void stage_tiles(const uint32_t* __restrict__ Xb,
                                            const uint32_t* __restrict__ Wb,
                                            __bf16* tileA, __bf16* tileB,
                                            int gmBase, int gnBase, int kk,
                                            int tid) {
#pragma unroll
  for (int j = 0; j < 8; ++j) {            // A: 256 rows * 8 x 16B
    const int idx = j * 256 + tid;
    const int row = idx >> 3;
    const int c   = idx & 7;
    copy16_to_lds(Xb + ((size_t)(gmBase + row) * K_TOT + kk) / 2 + c * 4,
                  tileA + row * LDK + c * 8);
  }
#pragma unroll
  for (int j = 0; j < 4; ++j) {            // B: 128 rows * 8 x 16B
    const int idx = j * 256 + tid;
    const int row = idx >> 3;
    const int c   = idx & 7;
    copy16_to_lds(Wb + ((size_t)(gnBase + row) * K_TOT + kk) / 2 + c * 4,
                  tileB + row * LDK + c * 8);
  }
}

// ---------------------------------------------------------------------------
// Per-wave 64x64 compute from one LDS buffer: 2 k-substeps x 16 WMMA.
// ---------------------------------------------------------------------------
__device__ __forceinline__ void compute_tiles(const __bf16* tileA,
                                              const __bf16* tileB,
                                              v8f acc[4][4], int waveM,
                                              int waveN, int r, int hi) {
#pragma unroll
  for (int ks = 0; ks < BK; ks += 32) {
    v16bf afrag[4], bfrag[4];
#pragma unroll
    for (int mi = 0; mi < 4; ++mi) {
      // A layout: lanes 0-15 -> K {0..7,16..23}; lanes 16-31 -> {8..15,24..31}
      const __bf16* p = tileA + (waveM * 64 + mi * 16 + r) * LDK + ks + hi * 8;
      const v8bf lo = *(const v8bf*)p;
      const v8bf hb = *(const v8bf*)(p + 16);
#pragma unroll
      for (int e = 0; e < 8; ++e) {
        afrag[mi][e]     = lo[e];
        afrag[mi][8 + e] = hb[e];
      }
    }
#pragma unroll
    for (int ni = 0; ni < 4; ++ni) {
      // B layout: lanes 0-15 -> K {0..7,16..23}; lanes 16-31 -> {8..15,24..31}
      const __bf16* p = tileB + (waveN * 64 + ni * 16 + r) * LDK + ks + hi * 8;
      const v8bf lo = *(const v8bf*)p;
      const v8bf hb = *(const v8bf*)(p + 16);
#pragma unroll
      for (int e = 0; e < 8; ++e) {
        bfrag[ni][e]     = lo[e];
        bfrag[ni][8 + e] = hb[e];
      }
    }
#pragma unroll
    for (int mi = 0; mi < 4; ++mi)
#pragma unroll
      for (int ni = 0; ni < 4; ++ni)
        acc[mi][ni] = __builtin_amdgcn_wmma_f32_16x16x32_bf16(
            false, afrag[mi], false, bfrag[ni], (short)0, acc[mi][ni], false,
            false);
  }
}

// ---------------------------------------------------------------------------
// Pass 2: GEMM  out[m][n] = sum_k xbf[m][k] * Wbf[n][k] + bias[n]
// WG tile 256x128, 8 waves, double-buffered LDS, one barrier per K-step.
// Ping-pong buffer selected by integer offset (no LDS pointer arrays -> no
// addrspacecast static initializers).
// ---------------------------------------------------------------------------
__global__ __launch_bounds__(256)
void anyprec_gemm_kernel(const uint32_t* __restrict__ Xb,
                         const uint32_t* __restrict__ Wb,
                         const float* __restrict__ bias,
                         float* __restrict__ out) {
  extern __shared__ __bf16 smem[]; // 2 x [BM+BN][LDK]

  const int tid   = threadIdx.x;
  const int lane  = tid & 31;
  const int wave  = tid >> 5;
  const int waveM = wave >> 1;
  const int waveN = wave & 1;
  const int r     = lane & 15;
  const int hi    = lane >> 4;

  const int gmBase = blockIdx.y * BM;
  const int gnBase = blockIdx.x * BN;

  v8f acc[4][4];
#pragma unroll
  for (int mi = 0; mi < 4; ++mi)
#pragma unroll
    for (int ni = 0; ni < 4; ++ni)
#pragma unroll
      for (int e = 0; e < 8; ++e) acc[mi][ni][e] = 0.0f;

  constexpr int NIT = K_TOT / BK; // 64

  // Prologue: stage tile 0 into buffer 0.
  stage_tiles(Xb, Wb, smem, smem + BM * LDK, gmBase, gnBase, 0, tid);

  for (int i = 0; i < NIT; ++i) {
    const int curOff = (i & 1) * BUF_ELEMS;
    const int nxtOff = BUF_ELEMS - curOff;
    wait_async_all();      // this wave's async writes into cur buf are done
    __syncthreads();       // everyone's are; next-buf readers are also done
    if (i + 1 < NIT)
      stage_tiles(Xb, Wb, smem + nxtOff, smem + nxtOff + BM * LDK, gmBase,
                  gnBase, (i + 1) * BK, tid);
    compute_tiles(smem + curOff, smem + curOff + BM * LDK, acc, waveM, waveN,
                  r, hi);
  }

  // Epilogue: C/D layout lane = N + 16*(M>=8), vgpr = M%8.
#pragma unroll
  for (int mi = 0; mi < 4; ++mi) {
#pragma unroll
    for (int ni = 0; ni < 4; ++ni) {
      const int gm = gmBase + waveM * 64 + mi * 16 + hi * 8;
      const int gn = gnBase + waveN * 64 + ni * 16 + r;
      const float bv = bias[gn];
#pragma unroll
      for (int v = 0; v < 8; ++v)
        out[(size_t)(gm + v) * N_TOT + gn] = acc[mi][ni][v] + bv;
    }
  }
}

// ---------------------------------------------------------------------------
// Host launcher. d_ws layout: [W bf16: 32 MiB][x bf16: 64 MiB] = 96 MiB.
// ---------------------------------------------------------------------------
extern "C" void kernel_launch(void* const* d_in, const int* in_sizes, int n_in,
                              void* d_out, int out_size, void* d_ws,
                              size_t ws_size, hipStream_t stream) {
  const float* x    = (const float*)d_in[0];
  const int*   qw   = (const int*)d_in[1];
  const float* lut  = (const float*)d_in[2];
  const float* bias = (const float*)d_in[3];
  float*       out  = (float*)d_out;

  uint32_t* Wbf = (uint32_t*)d_ws;
  uint32_t* Xbf = (uint32_t*)((char*)d_ws + (size_t)N_TOT * K_TOT * 2);

  anyprec_dequant_kernel<<<(N_TOT * NWORDS) / 256, 256, 0, stream>>>(qw, lut,
                                                                     Wbf);
  x_to_bf16_kernel<<<(M_TOT * K_TOT / 8) / 256, 256, 0, stream>>>(x, Xbf);

  dim3 grid(N_TOT / BN, M_TOT / BM);
  const size_t ldsBytes = 2 * (size_t)BUF_ELEMS * sizeof(__bf16); // 110592
  anyprec_gemm_kernel<<<grid, 256, ldsBytes, stream>>>(Xbf, Wbf, bias, out);
}
